// SamplerModel_22857815949524
// MI455X (gfx1250) — matrix operand order repaired and checked by reading, not compiled, not run on hardware
//
#include <hip/hip_runtime.h>

// ---- CDNA5 (gfx1250) wave32 WMMA types ----
typedef __attribute__((ext_vector_type(16))) __bf16 v16bf;
typedef __attribute__((ext_vector_type(8)))  float  v8f;
typedef __attribute__((ext_vector_type(4)))  float  v4f;

#define TOKENS   32768
#define DMODEL   1024
#define NEXP     64
#define NLOG     128              // 2*NEXP router outputs
#define TOPK     8
#define KCHUNK   32               // K per v_wmma_f32_16x16x32_bf16
#define KSUPER   256              // K staged into LDS per pass
#define NSUPER   (DMODEL / KSUPER)    // 4
#define KCPS     (KSUPER / KCHUNK)    // 8 chunks per superchunk
#define ROWS_PER_BLOCK 128        // 8 waves * 16-row M-tile
#define LDS_HALF (KCPS * 8 * 32 * 32) // 64 KB per bf16 plane (hi / lo)
#define LDS_TOTAL (2 * LDS_HALF)      // 128 KB dynamic LDS

__global__ __launch_bounds__(256, 2)
void moe_router_topk(const float* __restrict__ X, const float* __restrict__ W,
                     const float* __restrict__ NZ, const int* __restrict__ SKIP,
                     float* __restrict__ OUTP, float* __restrict__ OUTI)
{
  extern __shared__ char smem[];
  const int tid  = threadIdx.x;
  const int wave = tid >> 5;
  const int lane = tid & 31;
  const int h    = lane >> 4;      // K-half per 16-bit A/B fragment layout
  const int lm   = lane & 15;      // M (A rows) / N (B,C cols) within tile
  const int rowBase = blockIdx.x * ROWS_PER_BLOCK + wave * 16;

  v8f acc[8];
#pragma unroll
  for (int n = 0; n < 8; ++n)
#pragma unroll
    for (int r = 0; r < 8; ++r) acc[n][r] = 0.0f;

  for (int s = 0; s < NSUPER; ++s) {
    __syncthreads();
    const int k0 = s * KSUPER;
    // ---- cooperative stage: W[k0..k0+255, 0..127] -> LDS as bf16 hi/lo,
    //      pre-swizzled into the 32x16 B-fragment layout (ISA 7.12.2).
    //      Each unit = 4 consecutive K of one column -> one b64 LDS store.
    for (int i = 0; i < 32; ++i) {
      const int u  = tid + (i << 8);          // 8192 units
      const int nn = u & 127;                 // column 0..127
      const int kq = u >> 7;                  // K quad 0..63
      const int kc = kq >> 3;                 // chunk 0..7
      const int hh = (kq >> 1) & 1;           // K-half within chunk
      const int L  = hh * 16 + (nn & 15);     // owning lane
      const int fb = (((kc << 3) + (nn >> 4)) * 32 + L) * 32 + ((kq & 1) << 3) + (((kq >> 2) & 1) << 4);
      const float* gp = W + (size_t)(k0 + (kq << 2)) * NLOG + nn;
      union { __bf16 b[4]; unsigned long long q; } hp, lp;
#pragma unroll
      for (int t = 0; t < 4; ++t) {
        const float  x  = gp[t * NLOG];
        const __bf16 xh = (__bf16)x;          // hi (RNE)
        hp.b[t] = xh;
        lp.b[t] = (__bf16)(x - (float)xh);    // lo residual
      }
      *(unsigned long long*)(smem + fb)            = hp.q;
      *(unsigned long long*)(smem + LDS_HALF + fb) = lp.q;
    }
    __syncthreads();
    // ---- GEMM: 3 bf16 WMMAs per tile emulate fp32 (hi*hi + lo*hi + hi*lo)
    for (int kc = 0; kc < KCPS; ++kc) {
      // A fragment: lane holds row lm, K = h*8+{0..7} and 16+h*8+{0..7}
      const float* ap = X + (size_t)(rowBase + lm) * DMODEL + k0 + kc * KCHUNK + h * 8;
      const v4f a0 = *(const v4f*)(ap);
      const v4f a1 = *(const v4f*)(ap + 4);
      const v4f a2 = *(const v4f*)(ap + 16);
      const v4f a3 = *(const v4f*)(ap + 20);
      v16bf ah, al;
#pragma unroll
      for (int e = 0; e < 4; ++e) {
        float x; __bf16 xh;
        x = a0[e]; xh = (__bf16)x; ah[e]      = xh; al[e]      = (__bf16)(x - (float)xh);
        x = a1[e]; xh = (__bf16)x; ah[4 + e]  = xh; al[4 + e]  = (__bf16)(x - (float)xh);
        x = a2[e]; xh = (__bf16)x; ah[8 + e]  = xh; al[8 + e]  = (__bf16)(x - (float)xh);
        x = a3[e]; xh = (__bf16)x; ah[12 + e] = xh; al[12 + e] = (__bf16)(x - (float)xh);
      }
#pragma unroll
      for (int n = 0; n < 8; ++n) {
        const int fb = (((kc << 3) + n) * 32 + lane) * 32;
        const v16bf bh = *(const v16bf*)(smem + fb);
        const v16bf bl = *(const v16bf*)(smem + LDS_HALF + fb);
        acc[n] = __builtin_amdgcn_wmma_f32_16x16x32_bf16(false, ah, false, bh,
                                                         (short)0, acc[n], false, false);
        acc[n] = __builtin_amdgcn_wmma_f32_16x16x32_bf16(false, al, false, bh,
                                                         (short)0, acc[n], false, false);
        acc[n] = __builtin_amdgcn_wmma_f32_16x16x32_bf16(false, ah, false, bl,
                                                         (short)0, acc[n], false, false);
      }
    }
  }

  // ---- epilogue: spill C fragments (lane holds rows r+8h, col n*16+lm) to LDS
  __syncthreads();
  float* lds_log = (float*)smem + wave * (16 * NLOG);
#pragma unroll
  for (int n = 0; n < 8; ++n)
#pragma unroll
    for (int r = 0; r < 8; ++r)
      lds_log[(r + 8 * h) * NLOG + n * 16 + lm] = acc[n][r];
  __syncthreads();

  // ---- per-token noisy softmax + masked top-8 (one token per lane 0..15)
  if (lane < 16) {
    const int token = rowBase + lane;
    float* lg = lds_log + lane * NLOG;            // [0..63]=logits [64..127]=raw_std
    const float* nz = NZ   + (size_t)token * NEXP;
    const int*   sm = SKIP + (size_t)token * NEXP;
    float mx = -3.0e38f;
    for (int e = 0; e < NEXP; ++e) {
      const float raw = lg[NEXP + e];
      const float sd  = 1.0f / (1.0f + __expf(-raw)) + 0.01f;   // sigmoid + eps
      const float v   = lg[e] + nz[e] * sd;
      lg[e] = v;
      mx = fmaxf(mx, v);
    }
    float sum = 0.0f;
    for (int e = 0; e < NEXP; ++e) {
      const float ex = __expf(lg[e] - mx);
      lg[e] = ex; sum += ex;
    }
    const float inv = 1.0f / sum;
    for (int e = 0; e < NEXP; ++e)                 // softmax then mask (ref order)
      lg[e] = (sm[e] != 0) ? lg[e] * inv : 0.0f;

    float tsum = 0.0f;
    float tp[TOPK]; int ti[TOPK];
#pragma unroll
    for (int k = 0; k < TOPK; ++k) {               // strict > => lowest-index ties
      float best = -1.0f; int bi = 0;
      for (int e = 0; e < NEXP; ++e) {
        const float p = lg[e];
        if (p > best) { best = p; bi = e; }
      }
      tp[k] = best; ti[k] = bi; tsum += best;
      lg[bi] = -2.0f;                              // never re-selected (< -1 init)
    }
    const float rn = 1.0f / (tsum + 1e-6f);
#pragma unroll
    for (int k = 0; k < TOPK; ++k) {
      OUTP[(size_t)token * TOPK + k] = tp[k] * rn;
      OUTI[(size_t)token * TOPK + k] = (float)ti[k];
    }
  }
}

extern "C" void kernel_launch(void* const* d_in, const int* in_sizes, int n_in,
                              void* d_out, int out_size, void* d_ws, size_t ws_size,
                              hipStream_t stream) {
  (void)in_sizes; (void)n_in; (void)d_ws; (void)ws_size; (void)out_size;
  const float* X  = (const float*)d_in[0];   // [32768,1024]
  const float* W  = (const float*)d_in[1];   // [1024,128]
  const float* NZ = (const float*)d_in[2];   // [32768,64]
  const int*   SK = (const int*)  d_in[3];   // [32768,64]
  float* outp = (float*)d_out;                         // selected probs [T,8]
  float* outi = outp + (size_t)TOKENS * TOPK;          // indices (as float) [T,8]

  (void)hipFuncSetAttribute((const void*)moe_router_topk,
                            hipFuncAttributeMaxDynamicSharedMemorySize, LDS_TOTAL);
  dim3 grid(TOKENS / ROWS_PER_BLOCK);  // 256 blocks
  dim3 block(256);                     // 8 wave32s
  moe_router_topk<<<grid, block, LDS_TOTAL, stream>>>(X, W, NZ, SK, outp, outi);
}
